// IEBINS_19207093748126
// MI455X (gfx1250) — matrix operand order repaired and checked by reading, not compiled, not run on hardware
//
#include <hip/hip_runtime.h>
#include <hip/hip_bf16.h>

// ---------------------------------------------------------------------------
// IEBins iterative depth pipeline for MI455X (gfx1250, wave32, WMMA + TDM)
// Implicit-GEMM convolutions with v_wmma_f32_16x16x32_f16; B tiles staged to
// LDS by the Tensor Data Mover (tensor_load_to_lds) on interior tiles, manual
// zero-padded staging on image borders.
// ---------------------------------------------------------------------------

#ifndef __has_builtin
#define __has_builtin(x) 0
#endif

#if __has_builtin(__builtin_amdgcn_tensor_load_to_lds)
#define IEB_HAS_TDM 1
#endif

#if __has_builtin(__builtin_amdgcn_s_wait_tensorcnt)
#define IEB_TWAIT0() __builtin_amdgcn_s_wait_tensorcnt(0)
#else
#define IEB_TWAIT0() asm volatile("s_wait_tensorcnt 0x0" ::: "memory")
#endif

typedef __attribute__((ext_vector_type(16))) _Float16 v16h;
typedef __attribute__((ext_vector_type(8)))  _Float16 v8h;
typedef __attribute__((ext_vector_type(8)))  float    v8f;
typedef __attribute__((ext_vector_type(4)))  unsigned u32x4;
typedef __attribute__((ext_vector_type(4)))  int      i32x4;
typedef __attribute__((ext_vector_type(8)))  int      i32x8;

static constexpr int BATCH = 2;
static constexpr int HGT   = 120;
static constexpr int WID   = 160;
static constexpr int HW    = HGT * WID;
static constexpr int N     = BATCH * HW;                // 38400 pixels
static constexpr int TILES = BATCH * HGT * (WID / 16);  // 2400 workgroups
static constexpr int BN    = 16;                        // bins

// ---------------------------------------------------------------------------
// TDM: load one 2D tile (32 rows x 16 f16, row stride N elements) into LDS.
// Descriptor per CDNA5 ISA:
//   group0: count=1 | lds_addr | global_addr(57b) | type=2
//   group1: data_size=1(2B), tensor_dim0/1=0x7fffffff (no OOB clip),
//           tile_dim0=16, tile_dim1=32, tensor_dim0_stride=N
__device__ __forceinline__ void tdm_load_tile(const _Float16* gsrc, unsigned ldsoff)
{
#ifdef IEB_HAS_TDM
    unsigned long long ga = (unsigned long long)(uintptr_t)gsrc;
    u32x4 g0 = { 1u,                                    // count=1, user descriptor
                 ldsoff,                                // LDS byte address
                 (unsigned)(ga & 0xffffffffu),
                 (unsigned)((ga >> 32) & 0x01ffffffu) | 0x80000000u };  // type=2
    i32x8 g1 = { (int)0x00010000u,                      // data_size=2B, mask=0
                 (int)0xffff0000u,                      // tensor_dim0[15:0]<<16
                 (int)0xffff7fffu,                      // td0[31:16] | td1[15:0]<<16
                 (int)0x00107fffu,                      // td1[31:16] | tile_dim0=16
                 32,                                    // tile_dim1=32 rows
                 N,                                     // tensor_dim0_stride[31:0]
                 0, 0 };                                // stride hi / dim1 stride
    i32x4 gz = { 0, 0, 0, 0 };
#if defined(__clang_major__) && __clang_major__ >= 23
    i32x8 gz8 = { 0, 0, 0, 0, 0, 0, 0, 0 };
    __builtin_amdgcn_tensor_load_to_lds(g0, g1, gz, gz, gz8, 0);
#else
    __builtin_amdgcn_tensor_load_to_lds(g0, g1, gz, gz, 0);
#endif
#else
    (void)gsrc; (void)ldsoff;
#endif
}

// ---------------------------------------------------------------------------
// activation
template<int ACT>
__device__ __forceinline__ float actf(float v) {
    if constexpr (ACT == 1) {                 // relu
        return v > 0.f ? v : 0.f;
    } else if constexpr (ACT == 2) {          // sigmoid
        return 1.f / (1.f + __expf(-v));
    } else if constexpr (ACT == 3) {          // tanh
        float e2 = __expf(2.f * v);
        return (e2 - 1.f) / (e2 + 1.f);
    } else {
        return v;
    }
}

// ---------------------------------------------------------------------------
// Implicit-GEMM conv: out[o][n] = act( sum_k Wp[o][k] * patch[k][n] + bias[o] )
//   activations: f16 channel-major [C][N]
//   weights:     f16 packed [O][Kpad], k = (ky*KW+kx)*C + ci, zero padded
//   one workgroup = 16 consecutive pixels of one image row, 8 waves
//   wave w computes o-blocks w and w+8 (16 out-channels each)
template<int C, int KH, int KW, int PH, int PW, int ACT>
__global__ __launch_bounds__(256)
void conv_wmma(const _Float16* __restrict__ in,
               const _Float16* __restrict__ Wp,
               const float*    __restrict__ bias,
               _Float16*       __restrict__ out,
               int O)
{
    constexpr int K    = C * KH * KW;
    constexpr int NCH  = (K + 31) / 32;
    constexpr int Kpad = NCH * 32;

    __shared__ _Float16 Bt[32 * 16];          // K-chunk x pixel tile, [k][p]

    const int tile = blockIdx.x;
    const int xw   = tile % (WID / 16);
    const int t2   = tile / (WID / 16);
    const int y0   = t2 % HGT;
    const int bb   = t2 / HGT;
    const int x0   = xw * 16;
    const int pixbase = bb * HW + y0 * WID + x0;

    const int tid  = threadIdx.x;
    const int lane = tid & 31;
    const int wave = tid >> 5;
    const int nob  = O >> 4;

    const int col  = lane & 15;               // pixel within tile / N column
    const int koff = (lane & 16) ? 8  : 0;    // A fragment K sub-offset
    const int bko  = (lane & 16) ? 16 : 0;    // unused; B fragment also uses koff
    const int mhi  = (lane & 16) ? 8  : 0;    // D row offset

    v8f acc0 = {};
    v8f acc1 = {};

    int tapky = 0, tapkx = 0, ci0 = 0;        // incremental tap state (C%32==0)

    for (int ch = 0; ch < NCH; ++ch) {
        __syncthreads();

        // ---- stage B tile: 32 K-values x 16 pixels ----
        if constexpr (C % 32 == 0) {
            // chunk lies entirely inside one filter tap -> uniform geometry
            const int yy = y0 + tapky - PH;
            const int xs = x0 + tapkx - PW;
            const bool rowok = (yy >= 0) && (yy < HGT);
            bool staged = false;
#ifdef IEB_HAS_TDM
            const bool interior = rowok && (xs >= 0) && (xs + 16 <= WID);
            if (interior) {                   // block-uniform branch
                if (wave == 0) {
                    tdm_load_tile(in + (size_t)ci0 * N + (size_t)bb * HW
                                     + (size_t)yy * WID + xs,
                                  (unsigned)(uintptr_t)(void*)Bt);
                    IEB_TWAIT0();
                }
                staged = true;
            }
#endif
            if (!staged) {                    // border: manual zero-padded staging
                for (int f = tid; f < 512; f += 256) {
                    int k  = f >> 4;
                    int p  = f & 15;
                    int xx = xs + p;
                    _Float16 v = (_Float16)0.f;
                    if (rowok && xx >= 0 && xx < WID)
                        v = in[(size_t)(ci0 + k) * N + bb * HW + yy * WID + xx];
                    Bt[f] = v;
                }
            }
        } else {
            // generic path (encd1: C=16, pure shifts)
            for (int f = tid; f < 512; f += 256) {
                int k  = f >> 4;
                int p  = f & 15;
                int kk = ch * 32 + k;
                _Float16 v = (_Float16)0.f;
                if (kk < K) {
                    int tap = kk / C;
                    int ci  = kk - tap * C;
                    int ky  = tap / KW;
                    int kx  = tap - ky * KW;
                    int yy  = y0 + ky - PH;
                    int xx  = x0 + p + kx - PW;
                    if (yy >= 0 && yy < HGT && xx >= 0 && xx < WID)
                        v = in[(size_t)ci * N + bb * HW + yy * WID + xx];
                }
                Bt[f] = v;
            }
        }
        __syncthreads();

        // ---- B fragment (shared by both o-blocks of this wave) ----
        union { v16h v; _Float16 e[16]; } bf;
#pragma unroll
        for (int j = 0; j < 16; ++j)
            bf.e[j] = Bt[(koff + (j & 7) + ((j >> 3) << 4)) * 16 + col];

        // ---- A fragments + WMMA ----
        if (wave < nob) {
            const _Float16* ap =
                Wp + (size_t)(wave * 16 + col) * Kpad + ch * 32 + koff;
            union { v16h v; v8h h[2]; } af;
            af.h[0] = *(const v8h*)ap;
            af.h[1] = *(const v8h*)(ap + 16);
            __builtin_prefetch(ap + 32, 0, 1);   // next K-chunk (global_prefetch_b8)
            acc0 = __builtin_amdgcn_wmma_f32_16x16x32_f16(
                false, af.v, false, bf.v, (short)0, acc0, false, false);
        }
        if (wave + 8 < nob) {
            const _Float16* ap =
                Wp + (size_t)((wave + 8) * 16 + col) * Kpad + ch * 32 + koff;
            union { v16h v; v8h h[2]; } af;
            af.h[0] = *(const v8h*)ap;
            af.h[1] = *(const v8h*)(ap + 16);
            acc1 = __builtin_amdgcn_wmma_f32_16x16x32_f16(
                false, af.v, false, bf.v, (short)0, acc1, false, false);
        }

        // ---- advance tap state ----
        if constexpr (C % 32 == 0) {
            ci0 += 32;
            if (ci0 >= C) {
                ci0 = 0;
                if (++tapkx == KW) { tapkx = 0; ++tapky; }
            }
        }
    }

    // ---- epilogue: bias + activation, store f16 [O][N] ----
    if (wave < nob) {
#pragma unroll
        for (int r = 0; r < 8; ++r) {
            int och = wave * 16 + r + mhi;
            float v = acc0[r] + bias[och];
            out[(size_t)och * N + pixbase + col] = (_Float16)actf<ACT>(v);
        }
    }
    if (wave + 8 < nob) {
#pragma unroll
        for (int r = 0; r < 8; ++r) {
            int och = (wave + 8) * 16 + r + mhi;
            float v = acc1[r] + bias[och];
            out[(size_t)och * N + pixbase + col] = (_Float16)actf<ACT>(v);
        }
    }
}

// ---------------------------------------------------------------------------
// Weight packing: OIHW f32 -> [O][Kpad] f16, k = (ky*KW+kx)*C + ci
__global__ void pack_w(const float* __restrict__ w, _Float16* __restrict__ wp,
                       int O, int C, int KH, int KW, int Kpad)
{
    int u = blockIdx.x * 256 + threadIdx.x;
    if (u >= O * Kpad) return;
    int o  = u / Kpad;
    int kk = u - o * Kpad;
    float v = 0.f;
    int K = C * KH * KW;
    if (kk < K) {
        int tap = kk / C;
        int ci  = kk - tap * C;
        int ky  = tap / KW;
        int kx  = tap - ky * KW;
        v = w[(((size_t)o * C + ci) * KH + ky) * KW + kx];
    }
    wp[u] = (_Float16)v;
}

// ---------------------------------------------------------------------------
// Layout conversions & GRU elementwise
__global__ void ctx_to_hx(const float* __restrict__ ctx, _Float16* __restrict__ dst)
{   // (B,192,H,W) -> [192][N] f16
    size_t u = (size_t)blockIdx.x * 256 + threadIdx.x;
    if (u >= (size_t)192 * N) return;
    int c  = (int)(u / N);
    int n  = (int)(u - (size_t)c * N);
    int b  = n / HW;
    int hw = n - b * HW;
    dst[u] = (_Float16)ctx[((size_t)b * 192 + c) * HW + hw];
}

__global__ void h_init(const float* __restrict__ gh, float* __restrict__ h)
{   // (B,128,H,W) -> [128][N] f32
    size_t u = (size_t)blockIdx.x * 256 + threadIdx.x;
    if (u >= (size_t)128 * N) return;
    int c  = (int)(u / N);
    int n  = (int)(u - (size_t)c * N);
    int b  = n / HW;
    int hw = n - b * HW;
    h[u] = gh[((size_t)b * 128 + c) * HW + hw];
}

__global__ void h_to_hx(const float* __restrict__ h, _Float16* __restrict__ hx)
{
    size_t u = (size_t)blockIdx.x * 256 + threadIdx.x;
    if (u >= (size_t)128 * N) return;
    hx[u] = (_Float16)h[u];
}

__global__ void rh_to_hx(const _Float16* __restrict__ r, const float* __restrict__ h,
                         _Float16* __restrict__ hx)
{
    size_t u = (size_t)blockIdx.x * 256 + threadIdx.x;
    if (u >= (size_t)128 * N) return;
    hx[u] = (_Float16)((float)r[u] * h[u]);
}

__global__ void gru_combine(const _Float16* __restrict__ z, const _Float16* __restrict__ q,
                            float* __restrict__ h, _Float16* __restrict__ hx)
{
    size_t u = (size_t)blockIdx.x * 256 + threadIdx.x;
    if (u >= (size_t)128 * N) return;
    float zv = (float)z[u];
    float qv = (float)q[u];
    float hv = (1.f - zv) * h[u] + zv * qv;
    h[u]  = hv;
    hx[u] = (_Float16)hv;
}

// ---------------------------------------------------------------------------
// Bin edge initialization: uniform bins in [mn, mx]
__global__ void init_bins(float* __restrict__ edg, float* __restrict__ cdf,
                          _Float16* __restrict__ cd16,
                          const int* __restrict__ mnp, const int* __restrict__ mxp)
{
    int n = blockIdx.x * 256 + threadIdx.x;
    if (n >= N) return;
    float mn = (float)*mnp;
    float mx = (float)*mxp;
    float step = (mx - mn) * (1.f / (float)BN);
    float prev = mn;
    edg[n] = mn;
    for (int i = 1; i <= BN; ++i) {
        float e = mn + step * (float)i;
        edg[i * N + n] = e;
        float c = 0.5f * (prev + e);
        cdf [(i - 1) * N + n] = c;
        cd16[(i - 1) * N + n] = (_Float16)c;
        prev = e;
    }
}

// ---------------------------------------------------------------------------
// softmax over bins, depth_r / uncertainty / classified depth, bin update
__global__ void stats_update(const _Float16* __restrict__ logit,   // [16][N]
                             float* __restrict__ edg,              // [17][N]
                             float* __restrict__ cdf,              // [16][N]
                             _Float16* __restrict__ cd16,          // [16][N]
                             float* __restrict__ out, int it,
                             const int* __restrict__ mnp, const int* __restrict__ mxp)
{
    int n = blockIdx.x * 256 + threadIdx.x;
    if (n >= N) return;
    float mn = (float)*mnp;
    float mx = (float)*mxp;

    float lg[BN], cdv[BN];
    float m = -1e30f;
#pragma unroll
    for (int i = 0; i < BN; ++i) {
        lg[i]  = (float)logit[i * N + n];
        cdv[i] = cdf[i * N + n];
        m = fmaxf(m, lg[i]);
    }
    float s = 0.f;
#pragma unroll
    for (int i = 0; i < BN; ++i) { lg[i] = __expf(lg[i] - m); s += lg[i]; }
    float inv = 1.f / s;
    float dr = 0.f;
#pragma unroll
    for (int i = 0; i < BN; ++i) dr += lg[i] * inv * cdv[i];
    float var = 0.f;
#pragma unroll
    for (int i = 0; i < BN; ++i) {
        float d = cdv[i] - dr;
        var += lg[i] * inv * d * d;
    }
    float unc = sqrtf(var);

    out[(0 * 6 + it) * N + n] = dr;            // pred_depths_r
    out[(2 * 6 + it) * N + n] = unc;           // uncertainty_maps

    int cnt = 0;
#pragma unroll
    for (int i = 1; i < BN; ++i) cnt += (dr >= edg[i * N + n]) ? 1 : 0;
    int label = (dr >= edg[BN * N + n]) ? 0 : cnt;
    out[(1 * 6 + it) * N + n] = cdv[label];    // pred_depths_c

    // update_bins: recenter on dr, width = unc, clip cumsum
    float start = fmaxf(dr - 0.5f * unc, mn);
    float step  = unc * (1.f / (float)BN);
    float prev  = fminf(fmaxf(start, mn), mx);
    edg[n] = prev;
    for (int i = 1; i <= BN; ++i) {
        float e = fminf(fmaxf(start + step * (float)i, mn), mx);
        edg[i * N + n] = e;
        float c = 0.5f * (prev + e);
        cdf [(i - 1) * N + n] = c;
        cd16[(i - 1) * N + n] = (_Float16)c;
        prev = e;
    }
}

// ---------------------------------------------------------------------------
static inline int kpad_of(int C, int KH, int KW) {
    return ((C * KH * KW + 31) / 32) * 32;
}

extern "C" void kernel_launch(void* const* d_in, const int* in_sizes, int n_in,
                              void* d_out, int out_size, void* d_ws, size_t ws_size,
                              hipStream_t stream)
{
    (void)in_sizes; (void)n_in; (void)out_size; (void)ws_size;

    const float* context = (const float*)d_in[1];
    const float* gru_h   = (const float*)d_in[2];
    const float* W[12] = {
        (const float*)d_in[3],  (const float*)d_in[5],  (const float*)d_in[7],
        (const float*)d_in[9],  (const float*)d_in[11], (const float*)d_in[13],
        (const float*)d_in[15], (const float*)d_in[17], (const float*)d_in[19],
        (const float*)d_in[21], (const float*)d_in[23], (const float*)d_in[25] };
    const float* Bi[12] = {
        (const float*)d_in[4],  (const float*)d_in[6],  (const float*)d_in[8],
        (const float*)d_in[10], (const float*)d_in[12], (const float*)d_in[14],
        (const float*)d_in[16], (const float*)d_in[18], (const float*)d_in[20],
        (const float*)d_in[22], (const float*)d_in[24], (const float*)d_in[26] };
    const int* mnp = (const int*)d_in[28];
    const int* mxp = (const int*)d_in[29];
    float* out = (float*)d_out;

    // ---- carve workspace ----
    char*  base = (char*)d_ws;
    size_t off  = 0;
    auto carve = [&](size_t bytes) -> void* {
        void* p = base + off;
        off = (off + bytes + 255) & ~(size_t)255;
        return p;
    };
    _Float16* HX   = (_Float16*)carve((size_t)576 * N * 2);  // [h | d | context]
    _Float16* A1   = (_Float16*)carve((size_t)128 * N * 2);
    _Float16* A2   = (_Float16*)carve((size_t)128 * N * 2);
    _Float16* Zg   = (_Float16*)carve((size_t)128 * N * 2);
    _Float16* Rg   = (_Float16*)carve((size_t)128 * N * 2);
    _Float16* Qg   = (_Float16*)carve((size_t)128 * N * 2);
    float*    Hst  = (float*)   carve((size_t)128 * N * 4);
    float*    EDG  = (float*)   carve((size_t)17  * N * 4);
    float*    CDF  = (float*)   carve((size_t)16  * N * 4);
    _Float16* CD16 = (_Float16*)carve((size_t)16  * N * 2);
    _Float16* LOG  = (_Float16*)carve((size_t)16  * N * 2);

    // conv configs (O, C, KH, KW) in W[] order
    const int cfg[12][4] = {
        {128,  16, 7, 7},  // encd1
        {128, 128, 3, 3},  // encd2
        {128, 128, 3, 3},  // encd3
        {256, 128, 3, 3},  // encd4
        {128, 576, 1, 5},  // convz1
        {128, 576, 1, 5},  // convr1
        {128, 576, 1, 5},  // convq1
        {128, 576, 5, 1},  // convz2
        {128, 576, 5, 1},  // convr2
        {128, 576, 5, 1},  // convq2
        {128, 128, 3, 3},  // ph1
        { 16, 128, 3, 3}   // ph2
    };
    _Float16* Wp[12];
    for (int i = 0; i < 12; ++i) {
        int O = cfg[i][0], C = cfg[i][1], KH = cfg[i][2], KW = cfg[i][3];
        int Kp = kpad_of(C, KH, KW);
        Wp[i] = (_Float16*)carve((size_t)O * Kp * 2);
        int tot = O * Kp;
        pack_w<<<(tot + 255) / 256, 256, 0, stream>>>(W[i], Wp[i], O, C, KH, KW, Kp);
    }

    const int g128 = (128 * N + 255) / 256;
    const int g192 = (192 * N + 255) / 256;
    const int gN   = (N + 255) / 256;

    // ---- one-time per launch setup ----
    ctx_to_hx<<<g192, 256, 0, stream>>>(context, HX + (size_t)384 * N);
    h_init   <<<g128, 256, 0, stream>>>(gru_h, Hst);
    init_bins<<<gN,   256, 0, stream>>>(EDG, CDF, CD16, mnp, mxp);

    // ---- 6 refinement iterations ----
    for (int it = 0; it < 6; ++it) {
        // ProjectionInputDepth
        conv_wmma<16, 7, 7, 3, 3, 1><<<TILES, 256, 0, stream>>>(CD16, Wp[0], Bi[0], A1, 128);
        conv_wmma<128, 3, 3, 1, 1, 1><<<TILES, 256, 0, stream>>>(A1, Wp[1], Bi[1], A2, 128);
        conv_wmma<128, 3, 3, 1, 1, 1><<<TILES, 256, 0, stream>>>(A2, Wp[2], Bi[2], A1, 128);
        conv_wmma<128, 3, 3, 1, 1, 1><<<TILES, 256, 0, stream>>>(A1, Wp[3], Bi[3],
                                                                 HX + (size_t)128 * N, 256);
        // SepConvGRU horizontal (1x5)
        h_to_hx<<<g128, 256, 0, stream>>>(Hst, HX);
        conv_wmma<576, 1, 5, 0, 2, 2><<<TILES, 256, 0, stream>>>(HX, Wp[4], Bi[4], Zg, 128);
        conv_wmma<576, 1, 5, 0, 2, 2><<<TILES, 256, 0, stream>>>(HX, Wp[5], Bi[5], Rg, 128);
        rh_to_hx<<<g128, 256, 0, stream>>>(Rg, Hst, HX);
        conv_wmma<576, 1, 5, 0, 2, 3><<<TILES, 256, 0, stream>>>(HX, Wp[6], Bi[6], Qg, 128);
        gru_combine<<<g128, 256, 0, stream>>>(Zg, Qg, Hst, HX);
        // SepConvGRU vertical (5x1)
        conv_wmma<576, 5, 1, 2, 0, 2><<<TILES, 256, 0, stream>>>(HX, Wp[7], Bi[7], Zg, 128);
        conv_wmma<576, 5, 1, 2, 0, 2><<<TILES, 256, 0, stream>>>(HX, Wp[8], Bi[8], Rg, 128);
        rh_to_hx<<<g128, 256, 0, stream>>>(Rg, Hst, HX);
        conv_wmma<576, 5, 1, 2, 0, 3><<<TILES, 256, 0, stream>>>(HX, Wp[9], Bi[9], Qg, 128);
        gru_combine<<<g128, 256, 0, stream>>>(Zg, Qg, Hst, HX);
        // PHead: relu(conv3x3) -> conv3x3 logits   (HX[0:128] == f16 copy of h)
        conv_wmma<128, 3, 3, 1, 1, 1><<<TILES, 256, 0, stream>>>(HX, Wp[10], Bi[10], A1, 128);
        conv_wmma<128, 3, 3, 1, 1, 0><<<TILES, 256, 0, stream>>>(A1, Wp[11], Bi[11], LOG, 16);
        // softmax, outputs, bin update
        stats_update<<<gN, 256, 0, stream>>>(LOG, EDG, CDF, CD16, out, it, mnp, mxp);
    }
}
